// T_att_1803886264336
// MI455X (gfx1250) — matrix operand order repaired and checked by reading, not compiled, not run on hardware
//
#include <hip/hip_runtime.h>

typedef __attribute__((ext_vector_type(16))) _Float16 v16h;
typedef __attribute__((ext_vector_type(8)))  _Float16 v8h;
typedef __attribute__((ext_vector_type(8)))  float    v8f;
typedef __attribute__((ext_vector_type(4)))  float    v4f;

#define B_ 16
#define T_ 2048
#define C_ 256
#define H_ 256

// ---------------------------------------------------------------------------
// Swizzle W (C x H, fp32 row-major) into WMMA B-fragment f16 layout:
//   dst[(hblk*256 + fs*32 + lane)*16 + i] = W[f][h]
//   lane: N = h = hblk*16 + lane%16 ; half i: f = fs*32 + (lane<16 ? i : i+16)
// ---------------------------------------------------------------------------
__global__ __launch_bounds__(32) void wswz_kernel(const float* __restrict__ W,
                                                  _Float16* __restrict__ dst) {
  int lane = threadIdx.x;
  int idx  = blockIdx.x * 32 + lane;        // 0..4095 = [hblk 16][fs 8][lane 32]
  int hblk = idx >> 8;
  int fs   = (idx >> 5) & 7;
  int h    = hblk * 16 + (lane & 15);
  int fbase = fs * 32 + ((lane < 16) ? 0 : 16);
  _Float16* o = dst + (size_t)idx * 16;
#pragma unroll
  for (int i = 0; i < 16; ++i)
    o[i] = (_Float16)W[(fbase + i) * H_ + h];
}

// ---------------------------------------------------------------------------
// Projection: 16 rows x 256 cols of X@W + b per wave using f16 WMMA.
// transposed==0 : dst row-major f16 [BT][H]   (for Q, K)
// transposed==1 : dst f16 [B][H][T]           (for V, i.e. V^T)
// ---------------------------------------------------------------------------
__global__ __launch_bounds__(32) void proj_kernel(const float* __restrict__ X,
                                                  const _Float16* __restrict__ Wswz,
                                                  const float* __restrict__ bias,
                                                  _Float16* __restrict__ dst,
                                                  int transposed) {
  int lane  = threadIdx.x;
  int rb    = blockIdx.x;                // row-block over B*T (2048 blocks)
  int rbase = rb * 16;
  int mrow  = rbase + (lane & 15);
  int sel   = (lane < 16) ? 0 : 8;

  // A-fragments of X (16x32 f16 each), K dim = C = 8 steps of 32
  v16h xf[8];
#pragma unroll
  for (int fs = 0; fs < 8; ++fs) {
    const float* p0 = X + (size_t)mrow * C_ + fs * 32 + 2 * sel;   // f runs: +0..7
    const float* p1 = p0 + 8;                                  //         +8..15
    v4f a0 = *(const v4f*)p0;
    v4f a1 = *(const v4f*)(p0 + 4);
    v4f b0 = *(const v4f*)p1;
    v4f b1 = *(const v4f*)(p1 + 4);
    v16h f;
#pragma unroll
    for (int i = 0; i < 4; ++i) {
      f[i]      = (_Float16)a0[i];
      f[i + 4]  = (_Float16)a1[i];
      f[i + 8]  = (_Float16)b0[i];
      f[i + 12] = (_Float16)b1[i];
    }
    xf[fs] = f;
  }

#pragma unroll 1
  for (int hblk = 0; hblk < 16; ++hblk) {
    v8f acc = {};
    const _Float16* wp = Wswz + ((size_t)hblk * 256 + lane) * 16;
#pragma unroll
    for (int fs = 0; fs < 8; ++fs) {
      v16h wf = *(const v16h*)(wp + (size_t)fs * 32 * 16);
      acc = __builtin_amdgcn_wmma_f32_16x16x32_f16(false, xf[fs], false, wf,
                                                   (short)0, acc, false, false);
    }
    int   h  = hblk * 16 + (lane & 15);
    float bv = bias[h];
    if (!transposed) {
      // C layout: lane=N=h, VGPR r -> row rbase + r + sel
#pragma unroll
      for (int r = 0; r < 8; ++r) {
        int row = rbase + r + sel;
        dst[(size_t)row * H_ + h] = (_Float16)(acc[r] + bv);
      }
    } else {
      int b     = rb >> 7;
      int tbase = (rb & 127) * 16 + sel;
      v8h pk;
#pragma unroll
      for (int r = 0; r < 8; ++r) pk[r] = (_Float16)(acc[r] + bv);
      *(v8h*)(dst + ((size_t)(b * H_ + h)) * T_ + tbase) = pk;
    }
  }
}

// ---------------------------------------------------------------------------
// Flash attention: 8 waves per workgroup (128 queries), K/V staged through
// double-buffered LDS so all 8 waves share each 32-key block (8x less L2
// traffic).  Per wave: S^T = K_blk x Q^T (softmax row = lane), then
// O^T += V^T x P^T.  Qh/Kh row-major f16 [BT][H]; Vt f16 [B][H][T].
// ---------------------------------------------------------------------------
__global__ __launch_bounds__(256) void attn_kernel(const _Float16* __restrict__ Qh,
                                                   const _Float16* __restrict__ Kh,
                                                   const _Float16* __restrict__ Vt,
                                                   float* __restrict__ out) {
  int tid  = threadIdx.x;
  int lane = tid & 31;
  int wave = tid >> 5;                 // 0..7
  int qblk = blockIdx.x * 8 + wave;    // 0..127
  int b    = blockIdx.y;               // 0..15
  int q16  = lane & 15;
  int hi   = (lane >= 16);
  int off  = hi ? 8 : 0;

  // K block: 32 keys x 256 feats, row-major (same layout as Kh rows)
  // V block: 256 h x 32 s       (same layout as Vt columns sliced)
  __shared__ _Float16 Ks[2][32 * 256];
  __shared__ _Float16 Vs[2][256 * 32];

  // Q^T B-fragments: lane = query, 16 contiguous feature halves
  size_t qrow = (size_t)(b * T_ + qblk * 16 + q16) * H_;
  v16h qf[8];
#pragma unroll
  for (int fs = 0; fs < 8; ++fs)
    qf[fs] = *(const v16h*)(Qh + qrow + fs * 32 + (hi ? 16 : 0));

  float m = -1e30f, l = 0.f;
  v8f ot[16] = {};

  // ---- cooperative staging: each thread moves 64B of K and 64B of V ----
  v8h kreg[4], vreg[4];
  const size_t kbase = (size_t)(b * T_) * H_;            // halves
  const size_t vbase = (size_t)(b * H_ + tid) * T_;      // row h = tid
  {
    const v8h* ksrc = (const v8h*)(Kh + kbase);          // block 0
    const v8h* vsrc = (const v8h*)(Vt + vbase);
#pragma unroll
    for (int i = 0; i < 4; ++i) {
      kreg[i] = ksrc[tid * 4 + i];
      vreg[i] = vsrc[i];
    }
  }

  int buf = 0;
#pragma unroll 1
  for (int sb = 0; sb < T_ / 32; ++sb) {
    // store staged block, then issue next block's global loads (overlap)
    {
      v8h* kd = (v8h*)Ks[buf];
      v8h* vd = (v8h*)(Vs[buf] + tid * 32);
#pragma unroll
      for (int i = 0; i < 4; ++i) kd[tid * 4 + i] = kreg[i];
#pragma unroll
      for (int i = 0; i < 4; ++i) vd[i] = vreg[i];
    }
    if (sb + 1 < T_ / 32) {
      const v8h* ksrc = (const v8h*)(Kh + kbase + (size_t)(sb + 1) * 32 * H_);
      const v8h* vsrc = (const v8h*)(Vt + vbase + (size_t)(sb + 1) * 32);
#pragma unroll
      for (int i = 0; i < 4; ++i) {
        kreg[i] = ksrc[tid * 4 + i];
        vreg[i] = vsrc[i];
      }
    }
    __syncthreads();

    const _Float16* ks = Ks[buf];
    const _Float16* vs = Vs[buf];

    // ---- S^T tiles from LDS: st0 = local keys 0..15, st1 = 16..31 ----
    v8f st0 = {}, st1 = {};
#pragma unroll
    for (int fs = 0; fs < 8; ++fs) {
      union { v16h v; v8h h[2]; } ka, kb;
      const _Float16* pa = ks + (q16)      * H_ + fs * 32 + 2 * off;
      const _Float16* pb = ks + (q16 + 16) * H_ + fs * 32 + 2 * off;
      ka.h[0] = *(const v8h*)pa;  ka.h[1] = *(const v8h*)(pa + 8);
      kb.h[0] = *(const v8h*)pb;  kb.h[1] = *(const v8h*)(pb + 8);
      st0 = __builtin_amdgcn_wmma_f32_16x16x32_f16(false, ka.v, false, qf[fs],
                                                   (short)0, st0, false, false);
      st1 = __builtin_amdgcn_wmma_f32_16x16x32_f16(false, kb.v, false, qf[fs],
                                                   (short)0, st1, false, false);
    }

    // ---- online softmax (row = query = lane%16, replicated across halves) ----
    const float sc = 0.0625f;     // 1/sqrt(H)
    float s0[8], s1[8], bm = -1e30f;
#pragma unroll
    for (int r = 0; r < 8; ++r) {
      s0[r] = st0[r] * sc;
      s1[r] = st1[r] * sc;
      bm = fmaxf(bm, fmaxf(s0[r], s1[r]));
    }
    bm = fmaxf(bm, __shfl_xor(bm, 16, 32));
    float mn    = fmaxf(m, bm);
    float alpha = __expf(m - mn);
    float p0[8], p1[8], rs = 0.f;
#pragma unroll
    for (int r = 0; r < 8; ++r) {
      p0[r] = __expf(s0[r] - mn);
      p1[r] = __expf(s1[r] - mn);
      rs += p0[r] + p1[r];
    }
    rs += __shfl_xor(rs, 16, 32);
    l = l * alpha + rs;
    m = mn;

    // ---- build P^T B-fragment (lane = query, halves = 32 key slots) ----
    v16h pf;
#pragma unroll
    for (int r = 0; r < 8; ++r) {
      float x0 = __shfl_xor(p0[r], 16, 32);
      float x1 = __shfl_xor(p1[r], 16, 32);
      pf[r]     = (_Float16)(hi ? x1 : p0[r]);
      pf[r + 8] = (_Float16)(hi ? p1[r] : x0);
    }

    // ---- rescale O^T and accumulate O^T += V^T x P^T ----
#pragma unroll
    for (int t = 0; t < 16; ++t)
#pragma unroll
      for (int r = 0; r < 8; ++r) ot[t][r] *= alpha;

#pragma unroll
    for (int t = 0; t < 16; ++t) {
      union { v16h v; v8h h[2]; } vf;   // V^T A-frag: M = h, K = 32 key slots
      const _Float16* vp = vs + (t * 16 + q16) * 32 + 2 * off;
      vf.h[0] = *(const v8h*)vp;
      vf.h[1] = *(const v8h*)(vp + 8);
      ot[t] = __builtin_amdgcn_wmma_f32_16x16x32_f16(false, vf.v, false, pf,
                                                     (short)0, ot[t], false, false);
    }
    buf ^= 1;
  }

  // ---- epilogue: out[b][q][h] = O^T[h][q] / l ----
  float inv  = 1.f / l;
  size_t orow = (size_t)(b * T_ + qblk * 16 + q16) * H_;
  int hsel = hi ? 8 : 0;
#pragma unroll
  for (int t = 0; t < 16; ++t)
#pragma unroll
    for (int r = 0; r < 8; ++r)
      out[orow + t * 16 + r + hsel] = ot[t][r] * inv;
}

// ---------------------------------------------------------------------------
extern "C" void kernel_launch(void* const* d_in, const int* in_sizes, int n_in,
                              void* d_out, int out_size, void* d_ws, size_t ws_size,
                              hipStream_t stream) {
  const float* q  = (const float*)d_in[0];
  const float* k  = (const float*)d_in[1];
  const float* v  = (const float*)d_in[2];
  const float* Wq = (const float*)d_in[3];
  const float* bq = (const float*)d_in[4];
  const float* Wk = (const float*)d_in[5];
  const float* bk = (const float*)d_in[6];
  const float* Wv = (const float*)d_in[7];
  const float* bv = (const float*)d_in[8];
  float* out = (float*)d_out;

  char* ws = (char*)d_ws;
  _Float16* wsq = (_Float16*)ws;                 // 3 x 64K halves (swizzled W)
  _Float16* wsk = wsq + 65536;
  _Float16* wsv = wsk + 65536;
  size_t actOff = (3 * 65536 * sizeof(_Float16) + 255) & ~(size_t)255;
  _Float16* Qh = (_Float16*)(ws + actOff);       // [BT][H] f16, 16 MB
  _Float16* Kh = Qh + (size_t)B_ * T_ * H_;      // [BT][H] f16, 16 MB
  _Float16* Vt = Kh + (size_t)B_ * T_ * H_;      // [B][H][T] f16, 16 MB

  wswz_kernel<<<128, 32, 0, stream>>>(Wq, wsq);
  wswz_kernel<<<128, 32, 0, stream>>>(Wk, wsk);
  wswz_kernel<<<128, 32, 0, stream>>>(Wv, wsv);

  proj_kernel<<<2048, 32, 0, stream>>>(q, wsq, bq, Qh, 0);
  proj_kernel<<<2048, 32, 0, stream>>>(k, wsk, bk, Kh, 0);
  proj_kernel<<<2048, 32, 0, stream>>>(v, wsv, bv, Vt, 1);

  attn_kernel<<<dim3(16, B_), 256, 0, stream>>>(Qh, Kh, Vt, out);
}
